// MoEFFN_88545045774949
// MI455X (gfx1250) — compile-verified, matrix operand order checked by reading, not run on hardware
//
#include <hip/hip_runtime.h>
#include <hip/hip_bf16.h>

// Problem constants (B=2, S=2048)
#define T_TOK 4096
#define H_DIM 2048
#define I_DIM 5632
#define N_EXP 8

#define KSTEP 64      // K elements staged per tile
#define LDT   72      // LDS tile row stride in halfs (64 + 8 pad, keeps 16B align)

typedef __attribute__((ext_vector_type(16))) __bf16 v16bf;
typedef __attribute__((ext_vector_type(8)))  __bf16 v8bf;
typedef __attribute__((ext_vector_type(8)))  float  v8f;

union Frag16 { v16bf v; v8bf h[2]; };

// ---------------- workspace layout (bytes) ----------------
// hidden: 8192 compact rows + 32 dump rows (branchless epilogue overflow)
#define WS_COUNTS   0
#define WS_PROBSUM  32
#define WS_OFFSETS  64
#define WS_SLOTIDX  128
#define WS_SLOTW    (128 + 8*4096*4)
#define WS_HIDDEN   262400
#define HID_ROWS    8192

// fp32x8 -> bf16x8, single 16B LDS store
__device__ __forceinline__ void cvt_store8(__bf16* dst, float4 a, float4 b) {
    union { __bf16 h[8]; uint4 u; } t;
    t.h[0] = (__bf16)a.x; t.h[1] = (__bf16)a.y;
    t.h[2] = (__bf16)a.z; t.h[3] = (__bf16)a.w;
    t.h[4] = (__bf16)b.x; t.h[5] = (__bf16)b.y;
    t.h[6] = (__bf16)b.z; t.h[7] = (__bf16)b.w;
    *(uint4*)dst = t.u;
}

// ---------------------------------------------------------------------------
__global__ void moe_init(int* counts, float* probsum) {
    int t = threadIdx.x;
    if (t < N_EXP) { counts[t] = 0; probsum[t] = 0.f; }
}

__global__ void moe_zero_out(float* __restrict__ out) {
    size_t i = (size_t)blockIdx.x * blockDim.x + threadIdx.x;
    ((float4*)out)[i] = make_float4(0.f, 0.f, 0.f, 0.f);
}

// ---------------------------------------------------------------------------
// Router: logits -> softmax -> top2 -> scatter to expert lists + lb stats
__global__ __launch_bounds__(256) void moe_router(
    const float* __restrict__ x, const float* __restrict__ gate_w,
    int* __restrict__ counts, float* __restrict__ probsum,
    int* __restrict__ slot_idx, float* __restrict__ slot_w) {
    __shared__ float sGW[N_EXP * 1024];
    __shared__ float sPS[N_EXP];
    const int tid = threadIdx.x;
    const int t   = blockIdx.x * 256 + tid;
    if (tid < N_EXP) sPS[tid] = 0.f;

    float acc[N_EXP];
#pragma unroll
    for (int e = 0; e < N_EXP; e++) acc[e] = 0.f;

    for (int c = 0; c < 2; c++) {
        __syncthreads();
#pragma unroll
        for (int j = 0; j < 8; j++) {          // 8 x float4 per thread
            int elem = (tid + j * 256) * 4;
            int er = elem >> 10, col = elem & 1023;
            *(float4*)&sGW[elem] = *(const float4*)&gate_w[er * H_DIM + c * 1024 + col];
        }
        __syncthreads();
        const float* xr = x + (size_t)t * H_DIM + c * 1024;
#pragma unroll 4
        for (int h = 0; h < 1024; h++) {
            float xv = xr[h];
#pragma unroll
            for (int e = 0; e < N_EXP; e++)
                acc[e] = fmaf(xv, sGW[e * 1024 + h], acc[e]);
        }
    }

    float mx = acc[0];
#pragma unroll
    for (int e = 1; e < N_EXP; e++) mx = fmaxf(mx, acc[e]);
    float p[N_EXP], s = 0.f;
#pragma unroll
    for (int e = 0; e < N_EXP; e++) { p[e] = __expf(acc[e] - mx); s += p[e]; }
    float inv = 1.f / s;
#pragma unroll
    for (int e = 0; e < N_EXP; e++) p[e] *= inv;

#pragma unroll
    for (int e = 0; e < N_EXP; e++) atomicAdd(&sPS[e], p[e]);

    int i1 = 0; float p1 = p[0];
#pragma unroll
    for (int e = 1; e < N_EXP; e++) if (p[e] > p1) { p1 = p[e]; i1 = e; }
    int i2 = -1; float p2 = -1.f;
#pragma unroll
    for (int e = 0; e < N_EXP; e++)
        if (e != i1 && p[e] > p2) { p2 = p[e]; i2 = e; }
    float rs = 1.f / (p1 + p2);
    float w1 = p1 * rs, w2 = p2 * rs;

    int pos1 = atomicAdd(&counts[i1], 1);
    slot_idx[i1 * T_TOK + pos1] = t;  slot_w[i1 * T_TOK + pos1] = w1;
    int pos2 = atomicAdd(&counts[i2], 1);
    slot_idx[i2 * T_TOK + pos2] = t;  slot_w[i2 * T_TOK + pos2] = w2;

    __syncthreads();
    if (tid < N_EXP) atomicAdd(&probsum[tid], sPS[tid]);
}

// ---------------------------------------------------------------------------
__global__ void moe_finalize(const int* __restrict__ counts,
                             const float* __restrict__ probsum,
                             int* __restrict__ offsets,
                             float* __restrict__ out_lb) {
    if (threadIdx.x == 0) {
        int off = 0; float lb = 0.f;
        const float invT = 1.f / (float)T_TOK;
        for (int e = 0; e < N_EXP; e++) {
            offsets[e] = off; off += counts[e];
            lb += ((float)counts[e] * invT) * (probsum[e] * invT);
        }
        out_lb[0] = (float)N_EXP * lb;
    }
}

// ---------------------------------------------------------------------------
// GEMM1: hidden = silu(X_e Wg^T) * (X_e Wu^T)   (bf16 WMMA, f32 accum)
// block = 256 thr = 8 waves as 2(M) x 4(N); tile M=32 tokens, N=64 I-rows,
// K staged 64 at a time with register->LDS software pipeline.
__global__ __launch_bounds__(256) void moe_gemm1(
    const float* __restrict__ x,
    const float* __restrict__ w_gate, const float* __restrict__ w_up,
    const int* __restrict__ counts, const int* __restrict__ offsets,
    const int* __restrict__ slot_idx,
    unsigned short* __restrict__ hidden_raw) {
    const int e = blockIdx.y;
    const int i_base = blockIdx.x * 64;
    const int cnt = counts[e];
    if (cnt == 0) return;
    const int off = offsets[e];
    const int tid = threadIdx.x;
    const int lane = tid & 31;
    const int wm = (tid >> 5) & 1;
    const int wn = tid >> 6;

    __bf16* hidden = (__bf16*)hidden_raw;
    __shared__ __align__(16) __bf16 sX[32 * LDT];
    __shared__ __align__(16) __bf16 sG[64 * LDT];
    __shared__ __align__(16) __bf16 sU[64 * LDT];
    __shared__ int sTok[32];

    const float* wg_base = w_gate + ((size_t)e * I_DIM + i_base) * H_DIM;
    const float* wu_base = w_up   + ((size_t)e * I_DIM + i_base) * H_DIM;

    const int nl = lane & 15;
    const int khalf = (lane >> 4) << 3;   // 0 or 8: k-split per ISA frag layout
    const int srow = tid >> 3;            // staging row 0..31
    const int scol = (tid & 7) * 8;       // staging col 0..56

    for (int m0 = 0; m0 < cnt; m0 += 32) {
        __syncthreads();
        if (tid < 32) {
            int gm = m0 + tid;
            int cg = (gm < cnt) ? gm : (cnt - 1);   // clamp: no divergent guards
            sTok[tid] = slot_idx[e * T_TOK + cg];
        }
        __syncthreads();

        const float* xrow = x + (size_t)sTok[srow] * H_DIM + scol;
        const float* wg0  = wg_base + (size_t)srow        * H_DIM + scol;
        const float* wg1  = wg_base + (size_t)(srow + 32) * H_DIM + scol;
        const float* wu0  = wu_base + (size_t)srow        * H_DIM + scol;
        const float* wu1  = wu_base + (size_t)(srow + 32) * H_DIM + scol;

        float4 xA, xB, gA0, gB0, gA1, gB1, uA0, uB0, uA1, uB1;
        xA  = *(const float4*)(xrow);     xB  = *(const float4*)(xrow + 4);
        gA0 = *(const float4*)(wg0);      gB0 = *(const float4*)(wg0 + 4);
        gA1 = *(const float4*)(wg1);      gB1 = *(const float4*)(wg1 + 4);
        uA0 = *(const float4*)(wu0);      uB0 = *(const float4*)(wu0 + 4);
        uA1 = *(const float4*)(wu1);      uB1 = *(const float4*)(wu1 + 4);

        v8f accg = {}; v8f accu = {};
        for (int kk = 0; kk < H_DIM; kk += KSTEP) {
            cvt_store8(&sX[srow * LDT + scol],        xA,  xB);
            cvt_store8(&sG[srow * LDT + scol],        gA0, gB0);
            cvt_store8(&sG[(srow + 32) * LDT + scol], gA1, gB1);
            cvt_store8(&sU[srow * LDT + scol],        uA0, uB0);
            cvt_store8(&sU[(srow + 32) * LDT + scol], uA1, uB1);
            __syncthreads();

            int kn = (kk + KSTEP < H_DIM) ? (kk + KSTEP) : 0;
            xA  = *(const float4*)(xrow + kn);     xB  = *(const float4*)(xrow + kn + 4);
            gA0 = *(const float4*)(wg0 + kn);      gB0 = *(const float4*)(wg0 + kn + 4);
            gA1 = *(const float4*)(wg1 + kn);      gB1 = *(const float4*)(wg1 + kn + 4);
            uA0 = *(const float4*)(wu0 + kn);      uB0 = *(const float4*)(wu0 + kn + 4);
            uA1 = *(const float4*)(wu1 + kn);      uB1 = *(const float4*)(wu1 + kn + 4);

#pragma unroll
            for (int s = 0; s < 2; s++) {
                Frag16 a, bg, bu;
                const __bf16* pa = sX + (wm * 16 + nl) * LDT + s * 32 + khalf;
                a.h[0]  = *(const v8bf*)pa;       a.h[1]  = *(const v8bf*)(pa + 16);
                const __bf16* pg = sG + (wn * 16 + nl) * LDT + s * 32 + khalf;
                bg.h[0] = *(const v8bf*)pg;       bg.h[1] = *(const v8bf*)(pg + 16);
                const __bf16* pu = sU + (wn * 16 + nl) * LDT + s * 32 + khalf;
                bu.h[0] = *(const v8bf*)pu;       bu.h[1] = *(const v8bf*)(pu + 16);

                accg = __builtin_amdgcn_wmma_f32_16x16x32_bf16(
                    false, a.v, false, bg.v, (short)0, accg, false, false);
                accu = __builtin_amdgcn_wmma_f32_16x16x32_bf16(
                    false, a.v, false, bu.v, (short)0, accu, false, false);
            }
            __syncthreads();
        }

        // epilogue: silu(gate) * up -> compact bf16 hidden (branchless:
        // invalid rows of the last m-tile are redirected to dump rows)
        const int mh = (lane >> 4) << 3;
        const int colg = i_base + wn * 16 + nl;
#pragma unroll
        for (int i = 0; i < 8; i++) {
            int ml = i + mh;
            int gm = m0 + wm * 16 + ml;
            float g = accg[i], u = accu[i];
            float sg = __builtin_amdgcn_rcpf(1.f + __expf(-g));  // fast sigmoid
            float hv = g * sg * u;
            size_t row = (gm < cnt) ? (size_t)(off + gm)
                                    : (size_t)(HID_ROWS + wm * 16 + ml);
            hidden[row * I_DIM + colg] = (__bf16)hv;
        }
    }
}

// ---------------------------------------------------------------------------
// GEMM2: out[t,:] += w_t * hidden_row . Wd^T
// A frags loaded directly from global bf16 hidden; B staged fp32->bf16 via LDS.
// Branchless epilogue: invalid rows have wgt==0 and a clamped (valid) token,
// so the unconditional atomicAdd adds exactly 0.0f.
__global__ __launch_bounds__(256) void moe_gemm2(
    const unsigned short* __restrict__ hidden_raw,
    const float* __restrict__ w_down,
    const int* __restrict__ counts, const int* __restrict__ offsets,
    const int* __restrict__ slot_idx, const float* __restrict__ slot_w,
    float* __restrict__ out) {
    const int e = blockIdx.y;
    const int h_base = blockIdx.x * 64;
    const int cnt = counts[e];
    if (cnt == 0) return;
    const int off = offsets[e];
    const int tid = threadIdx.x;
    const int lane = tid & 31;
    const int wm = (tid >> 5) & 1;
    const int wn = tid >> 6;

    const __bf16* hidden = (const __bf16*)hidden_raw;
    __shared__ __align__(16) __bf16 sB[64 * LDT];
    __shared__ int   sTok[32];
    __shared__ float sWgt[32];

    const float* wd_base = w_down + ((size_t)e * H_DIM + h_base) * I_DIM;
    const int nl = lane & 15;
    const int khalf = (lane >> 4) << 3;
    const int srow = tid >> 3;
    const int scol = (tid & 7) * 8;

    for (int m0 = 0; m0 < cnt; m0 += 32) {
        __syncthreads();
        if (tid < 32) {
            int gm = m0 + tid;
            int cg = (gm < cnt) ? gm : (cnt - 1);
            sTok[tid] = slot_idx[e * T_TOK + cg];
            sWgt[tid] = (gm < cnt) ? slot_w[e * T_TOK + cg] : 0.f;
        }
        // per-lane A row (clamped: invalid M rows give zero-weighted outputs)
        int gmr = m0 + wm * 16 + nl;
        int arow = (gmr < cnt) ? gmr : (cnt - 1);
        const __bf16* pa_base = hidden + (size_t)(off + arow) * I_DIM + khalf;
        __syncthreads();

        const float* wd0 = wd_base + (size_t)srow        * I_DIM + scol;
        const float* wd1 = wd_base + (size_t)(srow + 32) * I_DIM + scol;
        float4 bA0, bB0, bA1, bB1;
        bA0 = *(const float4*)(wd0);  bB0 = *(const float4*)(wd0 + 4);
        bA1 = *(const float4*)(wd1);  bB1 = *(const float4*)(wd1 + 4);

        v8f acc = {};
        for (int kk = 0; kk < I_DIM; kk += KSTEP) {
            cvt_store8(&sB[srow * LDT + scol],        bA0, bB0);
            cvt_store8(&sB[(srow + 32) * LDT + scol], bA1, bB1);
            __syncthreads();

            int kn = (kk + KSTEP < I_DIM) ? (kk + KSTEP) : 0;
            bA0 = *(const float4*)(wd0 + kn);  bB0 = *(const float4*)(wd0 + kn + 4);
            bA1 = *(const float4*)(wd1 + kn);  bB1 = *(const float4*)(wd1 + kn + 4);

#pragma unroll
            for (int s = 0; s < 2; s++) {
                Frag16 a, b;
                const __bf16* pa = pa_base + kk + s * 32;
                a.h[0] = *(const v8bf*)pa;  a.h[1] = *(const v8bf*)(pa + 16);
                const __bf16* pb = sB + (wn * 16 + nl) * LDT + s * 32 + khalf;
                b.h[0] = *(const v8bf*)pb;  b.h[1] = *(const v8bf*)(pb + 16);

                acc = __builtin_amdgcn_wmma_f32_16x16x32_bf16(
                    false, a.v, false, b.v, (short)0, acc, false, false);
            }
            __syncthreads();
        }

        const int mh = (lane >> 4) << 3;
        const int colg = h_base + wn * 16 + nl;
#pragma unroll
        for (int i = 0; i < 8; i++) {
            int ml = i + mh;
            int   tok = sTok[wm * 16 + ml];
            float wgt = sWgt[wm * 16 + ml];
            atomicAdd(&out[(size_t)tok * H_DIM + colg], acc[i] * wgt);
        }
    }
}

// ---------------------------------------------------------------------------
extern "C" void kernel_launch(void* const* d_in, const int* in_sizes, int n_in,
                              void* d_out, int out_size, void* d_ws, size_t ws_size,
                              hipStream_t stream) {
    const float* x      = (const float*)d_in[0];
    const float* gate_w = (const float*)d_in[1];
    const float* w_gate = (const float*)d_in[2];
    const float* w_up   = (const float*)d_in[3];
    const float* w_down = (const float*)d_in[4];
    float* out = (float*)d_out;

    char* ws = (char*)d_ws;
    int*            counts   = (int*)(ws + WS_COUNTS);
    float*          probsum  = (float*)(ws + WS_PROBSUM);
    int*            offsets  = (int*)(ws + WS_OFFSETS);
    int*            slot_idx = (int*)(ws + WS_SLOTIDX);
    float*          slot_w   = (float*)(ws + WS_SLOTW);
    unsigned short* hidden   = (unsigned short*)(ws + WS_HIDDEN);

    moe_init<<<1, 32, 0, stream>>>(counts, probsum);
    moe_zero_out<<<(T_TOK * H_DIM / 4) / 256, 256, 0, stream>>>(out);
    moe_router<<<T_TOK / 256, 256, 0, stream>>>(x, gate_w, counts, probsum,
                                                slot_idx, slot_w);
    moe_finalize<<<1, 32, 0, stream>>>(counts, probsum, offsets,
                                       out + (size_t)T_TOK * H_DIM);
    dim3 g1(I_DIM / 64, N_EXP);   // 88 x 8
    moe_gemm1<<<g1, 256, 0, stream>>>(x, w_gate, w_up, counts, offsets,
                                      slot_idx, hidden);
    dim3 g2(H_DIM / 64, N_EXP);   // 32 x 8
    moe_gemm2<<<g2, 256, 0, stream>>>(hidden, w_down, counts, offsets,
                                      slot_idx, slot_w, out);
}
